// UnigramLM_84911503442815
// MI455X (gfx1250) — matrix operand order repaired and checked by reading, not compiled, not run on hardware
//
#include <hip/hip_runtime.h>
#include <hip/hip_bf16.h>

#define NEG   (-1.0e9f)
#define Tn    2048
#define Vn    32000
#define Ln    8
#define NCHUNK 64
#define CLEN   32   // NCHUNK * CLEN == Tn

typedef __attribute__((ext_vector_type(16))) _Float16 v16h;
typedef __attribute__((ext_vector_type(8)))  float    v8f;
typedef __attribute__((ext_vector_type(4)))  float    f4;

// ---------------------------------------------------------------------------
// Phase A: zero the 262MB posterior with 128-bit non-temporal stores
// (write-only streaming data; NT hint keeps it out of WGP$/L2 RT way).
// This is the roofline-dominant phase: 262MB / 23.3TB/s ~= 11us.
// ---------------------------------------------------------------------------
__global__ void zero_out_kernel(f4* __restrict__ out, long n4) {
    long i      = (long)blockIdx.x * blockDim.x + threadIdx.x;
    long stride = (long)gridDim.x * blockDim.x;
    f4 z = {0.f, 0.f, 0.f, 0.f};
    for (; i < n4; i += stride)
        __builtin_nontemporal_store(z, out + i);
}

// ---------------------------------------------------------------------------
// Phase B: w[t][j] = lpp[match_idx[t][j]] (or NEG); wrev[r][j] = w[r-j][j].
// ---------------------------------------------------------------------------
__global__ void build_w_kernel(const float* __restrict__ lpp,
                               const int*   __restrict__ midx,
                               float* __restrict__ w,
                               float* __restrict__ wrev) {
    int idx = blockIdx.x * blockDim.x + threadIdx.x;
    if (idx >= Tn * Ln) return;
    int t = idx >> 3, j = idx & 7;
    int mi = midx[idx];
    w[idx] = (mi >= 0) ? lpp[mi] : NEG;
    float wr = NEG;
    if (t >= j) {
        int mi2 = midx[(t - j) * Ln + j];
        if (mi2 >= 0) wr = lpp[mi2];
    }
    wrev[idx] = wr;
}

// ---------------------------------------------------------------------------
// Phase C: per-chunk 8x8 composite transition matrices (log semiring).
// One wave32 block per (direction, chunk); lanes 0..7 each propagate one
// basis column with state in registers; chunk's w rows staged in LDS.
// Sequential depth: CLEN=32 steps (vs 2048 for the naive scan).
// ---------------------------------------------------------------------------
__global__ __launch_bounds__(32)
void chunk_maps_kernel(const float* __restrict__ wrev,
                       const float* __restrict__ w,
                       float* __restrict__ fwdMat,
                       float* __restrict__ bwdMat) {
    __shared__ float tile[CLEN * Ln];
    int b = blockIdx.x, lane = threadIdx.x;
    bool is_fwd = (b < NCHUNK);
    int c = is_fwd ? b : (b - NCHUNK);

    for (int i = lane; i < CLEN * Ln; i += 32) {
        int s = i >> 3, j = i & 7;
        int row = is_fwd ? (c * CLEN + s) : (Tn - 1 - c * CLEN - s);
        tile[i] = is_fwd ? wrev[row * Ln + j] : w[row * Ln + j];
    }
    __syncthreads();

    if (lane < 8) {
        float u[8];
#pragma unroll
        for (int k = 0; k < 8; ++k) u[k] = (k == lane) ? 0.f : NEG;
        for (int s = 0; s < CLEN; ++s) {
            float xs[8];
            float m = -3.0e38f;
#pragma unroll
            for (int k = 0; k < 8; ++k) {
                xs[k] = u[k] + tile[s * 8 + k];
                m = fmaxf(m, xs[k]);
            }
            float sum = 0.f;
#pragma unroll
            for (int k = 0; k < 8; ++k) sum += __expf(xs[k] - m);
            float a = m + __logf(sum);
#pragma unroll
            for (int k = 7; k >= 1; --k) u[k] = u[k - 1];
            u[0] = a;
        }
        float* mat = is_fwd ? fwdMat : bwdMat;
#pragma unroll
        for (int k = 0; k < 8; ++k) mat[c * 64 + k * 8 + lane] = u[k];
    }
}

// ---------------------------------------------------------------------------
// Phase D: sequential prefix scan over the 64 chunk matrices, both directions
// at once as a block-diagonal 16x16 tile.  Log-matmul done as max-rescaled
// exp -> v_wmma_f32_16x16x32_f16 -> log (the LSE-matmul trick).  Column 0 /
// column 8 of the running prefix are the fwd/bwd chunk start-state vectors.
// Write-back is branchless: clamp acc to 1e-37 before log; a true zero maps
// to (mM+mR-85), which underflows to f16 0 at the next rescale, preserving
// the NEG semantics without per-element exec-mask branches.
// ---------------------------------------------------------------------------
__global__ __launch_bounds__(32)
void scan_prefix_kernel(const float* __restrict__ fwdMat,
                        const float* __restrict__ bwdMat,
                        float* __restrict__ fwdStart,
                        float* __restrict__ bwdStart) {
    __shared__ float R[256];   // running prefix, log domain, 16x16 row-major
    __shared__ float M[256];   // current chunk map, log domain
    int lane = threadIdx.x;
    int half_ = lane >> 4, rc = lane & 15;

    for (int i = lane; i < 256; i += 32)
        R[i] = ((i >> 4) == (i & 15)) ? 0.f : NEG;
    __syncthreads();

    for (int c = 0; c < NCHUNK; ++c) {
        // start states for chunk c come from the prefix BEFORE applying it
        if (lane < 8) {
            fwdStart[c * 8 + lane] = R[lane * 16 + 0];
            bwdStart[c * 8 + lane] = R[(8 + lane) * 16 + 8];
        }
        // assemble block-diagonal chunk map
        for (int i = lane; i < 256; i += 32) {
            int r = i >> 4, col = i & 15;
            float v = NEG;
            if (r < 8 && col < 8)        v = fwdMat[c * 64 + r * 8 + col];
            else if (r >= 8 && col >= 8) v = bwdMat[c * 64 + (r - 8) * 8 + (col - 8)];
            M[i] = v;
        }
        __syncthreads();

        // global maxes for exp-domain rescale
        float mM = -3.0e38f, mR = -3.0e38f;
        for (int i = lane; i < 256; i += 32) {
            mM = fmaxf(mM, M[i]);
            mR = fmaxf(mR, R[i]);
        }
#pragma unroll
        for (int off = 16; off; off >>= 1) {
            mM = fmaxf(mM, __shfl_xor(mM, off, 32));
            mR = fmaxf(mR, __shfl_xor(mR, off, 32));
        }

        // build WMMA fragments per documented 16-bit layouts (K=16..31 zero)
        v16h a, bfr;
#pragma unroll
        for (int v = 0; v < 8; ++v) {
            int Ka = (v < 4) ? (2 * v + 8 * half_) : (16 + 2 * (v - 4) + 8 * half_);
            float a0 = (Ka < 16)     ? __expf(M[rc * 16 + Ka]       - mM) : 0.f;
            float a1 = (Ka + 1 < 16) ? __expf(M[rc * 16 + Ka + 1]   - mM) : 0.f;
            a[2 * v]     = (_Float16)a0;
            a[2 * v + 1] = (_Float16)a1;
            int Kb = (v < 4) ? (2 * v + 8 * half_) : (16 + 2 * (v - 4) + 8 * half_);
            float b0 = (Kb < 16)     ? __expf(R[Kb * 16 + rc]       - mR) : 0.f;
            float b1 = (Kb + 1 < 16) ? __expf(R[(Kb + 1) * 16 + rc] - mR) : 0.f;
            bfr[2 * v]     = (_Float16)b0;
            bfr[2 * v + 1] = (_Float16)b1;
        }
        v8f acc = {};
        acc = __builtin_amdgcn_wmma_f32_16x16x32_f16(
            false, a, false, bfr, (short)0, acc, false, false);
        __syncthreads();

        // R_new = log(M_exp x R_exp) + mM + mR   (branchless epilogue)
        float scale = mM + mR;
#pragma unroll
        for (int r = 0; r < 8; ++r) {
            int row = r + 8 * half_;
            float cv = fmaxf(acc[r], 1.0e-37f);
            R[row * 16 + rc] = __logf(cv) + scale;
        }
        __syncthreads();
    }
}

// ---------------------------------------------------------------------------
// Phase E: replay each chunk from its true start state; lanes 0..7 hold the
// rolling state, LSE via width-8 shfl_xor, shift via shfl_up.  Writes
// alpha[0..T] / beta[0..T].
// ---------------------------------------------------------------------------
__global__ __launch_bounds__(32)
void chunk_replay_kernel(const float* __restrict__ wrev,
                         const float* __restrict__ w,
                         const float* __restrict__ fwdStart,
                         const float* __restrict__ bwdStart,
                         float* __restrict__ alpha,
                         float* __restrict__ beta) {
    __shared__ float tile[CLEN * Ln];
    int b = blockIdx.x, lane = threadIdx.x;
    bool is_fwd = (b < NCHUNK);
    int c = is_fwd ? b : (b - NCHUNK);

    for (int i = lane; i < CLEN * Ln; i += 32) {
        int s = i >> 3, j = i & 7;
        int row = is_fwd ? (c * CLEN + s) : (Tn - 1 - c * CLEN - s);
        tile[i] = is_fwd ? wrev[row * Ln + j] : w[row * Ln + j];
    }
    __syncthreads();

    int j = lane & 7;
    float sig = is_fwd ? fwdStart[c * 8 + j] : bwdStart[c * 8 + j];
    if (lane == 0 && c == 0) {
        if (is_fwd) alpha[0] = 0.f;
        else        beta[Tn] = 0.f;
    }

    for (int s = 0; s < CLEN; ++s) {
        float x = sig + tile[s * 8 + j];
        float m = x;
        m = fmaxf(m, __shfl_xor(m, 1, 8));
        m = fmaxf(m, __shfl_xor(m, 2, 8));
        m = fmaxf(m, __shfl_xor(m, 4, 8));
        float e = __expf(x - m);
        e += __shfl_xor(e, 1, 8);
        e += __shfl_xor(e, 2, 8);
        e += __shfl_xor(e, 4, 8);
        float a = m + __logf(e);
        if (lane == 0) {
            if (is_fwd) alpha[c * CLEN + s + 1] = a;
            else        beta[Tn - 1 - c * CLEN - s] = a;
        }
        float prev = __shfl_up(sig, 1, 8);
        sig = (j == 0) ? a : prev;
    }
}

// ---------------------------------------------------------------------------
// Phase F: scatter posteriors (16K atomics -- negligible).
// ---------------------------------------------------------------------------
__global__ void scatter_kernel(const int*   __restrict__ midx,
                               const float* __restrict__ w,
                               const float* __restrict__ alpha,
                               const float* __restrict__ beta,
                               float* __restrict__ P) {
    int idx = blockIdx.x * blockDim.x + threadIdx.x;
    if (idx >= Tn * Ln) return;
    int t = idx >> 3, j = idx & 7;
    int mi = midx[idx];
    if (mi < 0) return;
    float lp = alpha[t] + w[idx] + beta[t + j + 1] - alpha[Tn];
    atomicAdd(&P[(size_t)t * Vn + mi], __expf(lp));
}

// ---------------------------------------------------------------------------
extern "C" void kernel_launch(void* const* d_in, const int* in_sizes, int n_in,
                              void* d_out, int out_size, void* d_ws, size_t ws_size,
                              hipStream_t stream) {
    (void)in_sizes; (void)n_in; (void)out_size; (void)ws_size;
    const float* lpp  = (const float*)d_in[0];   // [V] log piece probs
    const int*   midx = (const int*)d_in[1];     // [T, L] match indices
    float* P = (float*)d_out;                    // [T, V]

    float* ws       = (float*)d_ws;
    float* w        = ws;                        // T*L
    float* wrev     = w        + Tn * Ln;        // T*L
    float* fwdMat   = wrev     + Tn * Ln;        // NCHUNK*64
    float* bwdMat   = fwdMat   + NCHUNK * 64;    // NCHUNK*64
    float* fwdStart = bwdMat   + NCHUNK * 64;    // NCHUNK*8
    float* bwdStart = fwdStart + NCHUNK * 8;     // NCHUNK*8
    float* alpha    = bwdStart + NCHUNK * 8;     // T+1
    float* beta     = alpha    + (Tn + 1);       // T+1

    long n4 = (long)Tn * Vn / 4;
    zero_out_kernel<<<4096, 256, 0, stream>>>((f4*)P, n4);
    build_w_kernel<<<(Tn * Ln + 255) / 256, 256, 0, stream>>>(lpp, midx, w, wrev);
    chunk_maps_kernel<<<2 * NCHUNK, 32, 0, stream>>>(wrev, w, fwdMat, bwdMat);
    scan_prefix_kernel<<<1, 32, 0, stream>>>(fwdMat, bwdMat, fwdStart, bwdStart);
    chunk_replay_kernel<<<2 * NCHUNK, 32, 0, stream>>>(wrev, w, fwdStart, bwdStart,
                                                       alpha, beta);
    scatter_kernel<<<(Tn * Ln + 255) / 256, 256, 0, stream>>>(midx, w, alpha, beta, P);
}
